// ISAHead_20418274525233
// MI455X (gfx1250) — matrix operand order repaired and checked by reading, not compiled, not run on hardware
//
#include <hip/hip_runtime.h>

// ---------------------------------------------------------------------------
// Types for gfx1250 WMMA (wave32, 16x16x32 f16 -> f32)
// ---------------------------------------------------------------------------
typedef __attribute__((ext_vector_type(16))) _Float16 v16h;
typedef __attribute__((ext_vector_type(8)))  float    v8f;
typedef __attribute__((ext_vector_type(4)))  float    f32x4;
typedef __attribute__((ext_vector_type(4)))  int      v4i;

union AFrag { v16h v; f32x4 q[2]; };
union HPack { f32x4 q; _Float16 h[8]; };

#define WMMA_F16(A,B,C) \
  __builtin_amdgcn_wmma_f32_16x16x32_f16(false,(A),false,(B),(short)0,(C),false,false)

// ---------------------------------------------------------------------------
// CDNA5 async global->LDS copy (ASYNCcnt-tracked), gated on builtin presence.
// Probe result: builtin exists with signature (int4 AS1*, int4 AS3*, Ii, Ii).
// ---------------------------------------------------------------------------
#if defined(__has_builtin)
#if __has_builtin(__builtin_amdgcn_global_load_async_to_lds_b128)
#define HAVE_ASYNC_LDS 1
#endif
#endif
#ifndef HAVE_ASYNC_LDS
#define HAVE_ASYNC_LDS 0
#endif

#if HAVE_ASYNC_LDS
typedef __attribute__((address_space(1))) v4i as1_v4i;
typedef __attribute__((address_space(3))) v4i as3_v4i;
#endif

__device__ __forceinline__ void copy16(const _Float16* g, _Float16* l)
{
#if HAVE_ASYNC_LDS
  __builtin_amdgcn_global_load_async_to_lds_b128((as1_v4i*)g, (as3_v4i*)l, 0, 0);
#else
  *(f32x4*)l = *(const f32x4*)g;
#endif
}

#if HAVE_ASYNC_LDS
template <int N>
__device__ __forceinline__ void wait_async_le()
{
#if __has_builtin(__builtin_amdgcn_s_wait_asynccnt)
  __builtin_amdgcn_s_wait_asynccnt(N);
#else
  asm volatile("s_wait_asynccnt %0" :: "n"(N) : "memory");
#endif
}
#endif

// ---------------------------------------------------------------------------
// fp32 [B,C,H,W] -> fp16 [token, C]  (token = b*HW + h*W + w), LDS transpose
// ---------------------------------------------------------------------------
__global__ void __launch_bounds__(256)
nchw_to_nc(const float* __restrict__ x, _Float16* __restrict__ out, int Cch, int HW)
{
  __shared__ _Float16 tile[32][33];
  int b   = blockIdx.z;
  int hw0 = blockIdx.x * 32;
  int c0  = blockIdx.y * 32;
  int tx = threadIdx.x & 31, ty = threadIdx.x >> 5;      // 32 x 8
  const float* src = x + (long)b * Cch * HW;
#pragma unroll
  for (int i = 0; i < 32; i += 8) {
    int c = c0 + ty + i;
    tile[ty + i][tx] = (_Float16)src[(long)c * HW + hw0 + tx];
  }
  __syncthreads();
  _Float16* dst = out + (long)b * HW * Cch;
#pragma unroll
  for (int i = 0; i < 32; i += 8) {
    int hw = hw0 + ty + i;
    dst[(long)hw * Cch + c0 + tx] = tile[tx][ty + i];
  }
}

// ---------------------------------------------------------------------------
// fp32 -> fp16 weight convert
// ---------------------------------------------------------------------------
__global__ void cvt_w(const float* __restrict__ in, _Float16* __restrict__ out, int n)
{
  int i = blockIdx.x * 256 + threadIdx.x;
  if (i < n) out[i] = (_Float16)in[i];
}

// w3 [O=512][I=1024][3][3] -> W3r [O][9216], k = (ky*3+kx)*1024 + i
__global__ void repack_w3(const float* __restrict__ w3, _Float16* __restrict__ W3r)
{
  int idx = blockIdx.x * 256 + threadIdx.x;          // < 512*9216
  if (idx >= 512 * 9216) return;
  int o = idx / 9216, k = idx % 9216;
  int t = k / 1024, i = k % 1024;
  int ky = t / 3, kx = t % 3;
  W3r[idx] = (_Float16)w3[(((long)o * 1024 + i) * 3 + ky) * 3 + kx];
}

// ---------------------------------------------------------------------------
// Bulk GEMM: Y[n,o] = act(scale[o]*(sum_c X[n,c]*W[o,c]) + shift[o])
// Block: 256 thr = 8 waves (2x4); block tile 64(M) x 256(N); wave tile 32x64.
// Double-buffered LDS staging of A (64x32) and B (256x32) tiles via
// async global->LDS (ASYNCcnt) when available.
// TRANS=true: per-group transposed store Y[g][o][GS] (for V^T).
// ---------------------------------------------------------------------------
#define LDT 40   // padded row stride (halves) for 32-wide K tiles

template <bool TRANS>
__global__ void __launch_bounds__(256)
gemm_f16(const _Float16* __restrict__ X, const _Float16* __restrict__ W,
         const float* __restrict__ scale, const float* __restrict__ shift,
         int relu, _Float16* __restrict__ Y, long ldy, int ocol_off,
         int Cin, int Cout, int GS)
{
  __shared__ _Float16 Asm[2][64 * LDT];    // 10.0 KB
  __shared__ _Float16 Bsm[2][256 * LDT];   // 40.0 KB
  int tid  = threadIdx.x;
  int lane = tid & 31, wv = tid >> 5;
  int wm = wv >> 2, wn = wv & 3;                         // 2(M) x 4(N)
  long rowBase = (long)blockIdx.y * 64;
  int  colBase = blockIdx.x * 256;
  int nl = lane & 15, hi = lane >> 4;
  int aK = hi * 16, bK = hi * 16;

  // copy-in indexing: 16B chunk per thread for A, 4 chunks for B
  int crow = tid >> 2, ccol = (tid & 3) * 8;             // crow 0..63
  const _Float16* aSrc = X + (rowBase + crow) * (long)Cin + ccol;
  const _Float16* bSrc = W + (long)(colBase + crow) * Cin + ccol;
  int cdst = crow * LDT + ccol;
  long bRow64 = 64L * Cin;

  auto stage = [&](int buf, int k0) {
    copy16(aSrc + k0, &Asm[buf][cdst]);
#pragma unroll
    for (int j = 0; j < 4; j++)
      copy16(bSrc + k0 + j * bRow64, &Bsm[buf][cdst + j * 64 * LDT]);
  };

  v8f acc[2][4] = {};
  int S = Cin >> 5;
  stage(0, 0);
  for (int s = 0; s < S; s++) {
    int cur = s & 1;
    if (s + 1 < S) {
      stage(cur ^ 1, (s + 1) << 5);
#if HAVE_ASYNC_LDS
      wait_async_le<5>();     // 5 ops/stage in flight for next stage
#endif
    } else {
#if HAVE_ASYNC_LDS
      wait_async_le<0>();
#endif
    }
    __syncthreads();
    AFrag af[2], bf[4];
#pragma unroll
    for (int sm = 0; sm < 2; sm++) {
      const _Float16* p = &Asm[cur][(wm * 32 + sm * 16 + nl) * LDT + aK];
      af[sm].q[0] = *(const f32x4*)p;
      af[sm].q[1] = *(const f32x4*)(p + 8);
    }
#pragma unroll
    for (int sn = 0; sn < 4; sn++) {
      const _Float16* p = &Bsm[cur][(wn * 64 + sn * 16 + nl) * LDT + bK];
      bf[sn].q[0] = *(const f32x4*)p;
      bf[sn].q[1] = *(const f32x4*)(p + 8);
    }
#pragma unroll
    for (int sm = 0; sm < 2; sm++)
#pragma unroll
      for (int sn = 0; sn < 4; sn++)
        acc[sm][sn] = WMMA_F16(af[sm].v, bf[sn].v, acc[sm][sn]);
    __syncthreads();
  }

  int  col0w = colBase + wn * 64;
  long row0w = rowBase + wm * 32;
#pragma unroll
  for (int sm = 0; sm < 2; sm++) {
#pragma unroll
    for (int sn = 0; sn < 4; sn++) {
      int o = col0w + sn * 16 + nl;
      float sc = scale ? scale[o] : 1.0f;
      float sh = shift ? shift[o] : 0.0f;
      long nb = row0w + sm * 16 + hi * 8;
      if (!TRANS) {
#pragma unroll
        for (int r = 0; r < 8; r++) {
          float v = acc[sm][sn][r] * sc + sh;
          if (relu) v = fmaxf(v, 0.0f);
          Y[(nb + r) * ldy + ocol_off + o] = (_Float16)v;
        }
      } else {
        long g = nb / GS, m0 = nb % GS;
        HPack pk;
#pragma unroll
        for (int r = 0; r < 8; r++) {
          float v = acc[sm][sn][r] * sc + sh;
          if (relu) v = fmaxf(v, 0.0f);
          pk.h[r] = (_Float16)v;
        }
        *(f32x4*)&Y[(g * Cout + o) * GS + m0] = pk.q;
      }
    }
  }
}

// ---------------------------------------------------------------------------
// sim[g][n][m] = rscale * sum_c Q[g*GS+n][c] * K[g*GS+m][c]
// Block tile 64x64; 8 waves as 4(M) x 2(N); wave tile 16x32.
// ---------------------------------------------------------------------------
__global__ void __launch_bounds__(256)
attn_sim(const _Float16* __restrict__ Q, const _Float16* __restrict__ Kx,
         float* __restrict__ S, int GS, int Cin, float rscale)
{
  int g = blockIdx.z;
  const _Float16* Qg = Q + (long)g * GS * Cin;
  const _Float16* Kg = Kx + (long)g * GS * Cin;
  float* Sg = S + (long)g * GS * GS;
  int lane = threadIdx.x & 31;
  int wv = threadIdx.x >> 5;
  int wm = wv >> 1, wn = wv & 1;
  int row0 = blockIdx.y * 64 + wm * 16;
  int col0 = blockIdx.x * 64 + wn * 32;
  int nl = lane & 15, hi = lane >> 4;
  int aK = hi * 16, bK = hi * 16;
  v8f acc0 = {}, acc1 = {};
  const _Float16* ap = Qg + (long)(row0 + nl) * Cin + aK;
  const _Float16* bp = Kg + (long)(col0 + nl) * Cin + bK;
  long bS = 16L * Cin;
  for (int k = 0; k < Cin; k += 32) {
    AFrag a, b0, b1;
    a.q[0]  = *(const f32x4*)(ap);      a.q[1]  = *(const f32x4*)(ap + 8);
    b0.q[0] = *(const f32x4*)(bp);      b0.q[1] = *(const f32x4*)(bp + 8);
    b1.q[0] = *(const f32x4*)(bp + bS); b1.q[1] = *(const f32x4*)(bp + bS + 8);
    acc0 = WMMA_F16(a.v, b0.v, acc0);
    acc1 = WMMA_F16(a.v, b1.v, acc1);
    ap += 32; bp += 32;
  }
#pragma unroll
  for (int r = 0; r < 8; r++) {
    int n = row0 + hi * 8 + r;
    Sg[(long)n * GS + col0 + nl]      = acc0[r] * rscale;
    Sg[(long)n * GS + col0 + 16 + nl] = acc1[r] * rscale;
  }
}

// ---------------------------------------------------------------------------
// Row softmax: f32 [rows][GS] -> f16 probabilities. One wave per row.
// ---------------------------------------------------------------------------
__global__ void __launch_bounds__(256)
softmax_rows(const float* __restrict__ S, _Float16* __restrict__ P, int GS)
{
  int row  = blockIdx.x * 8 + (threadIdx.x >> 5);
  int lane = threadIdx.x & 31;
  const float* sr = S + (long)row * GS;
  float mx = -1e30f;
  for (int i = lane; i < GS; i += 32) mx = fmaxf(mx, sr[i]);
#pragma unroll
  for (int off = 16; off; off >>= 1) mx = fmaxf(mx, __shfl_xor(mx, off, 32));
  float ebuf[8];
  float sum = 0.0f;
  int cnt = 0;
  for (int i = lane; i < GS; i += 32) {
    float e = __expf(sr[i] - mx);
    ebuf[cnt++] = e; sum += e;
  }
#pragma unroll
  for (int off = 16; off; off >>= 1) sum += __shfl_xor(sum, off, 32);
  float inv = 1.0f / sum;
  _Float16* pr = P + (long)row * GS;
  cnt = 0;
  for (int i = lane; i < GS; i += 32) pr[i] = (_Float16)(ebuf[cnt++] * inv);
}

// ---------------------------------------------------------------------------
// ctx[g*GS+n][c] = sum_m P[g][n][m] * Vt[g][c][m]
// ---------------------------------------------------------------------------
__global__ void __launch_bounds__(256)
attn_ctx(const _Float16* __restrict__ P, const _Float16* __restrict__ Vt,
         _Float16* __restrict__ Y, int GS, int Cch)
{
  int lane = threadIdx.x & 31;
  int wv = threadIdx.x >> 5;
  int wm = wv >> 2, wn = wv & 3;
  long row0 = (long)blockIdx.y * 64 + wm * 32;   // global token
  int  col0 = blockIdx.x * 128 + wn * 32;        // channel
  long g = row0 / GS;
  int  n0 = (int)(row0 % GS);
  const _Float16* Pg = P  + g * GS * GS;
  const _Float16* Vg = Vt + g * (long)Cch * GS;
  int nl = lane & 15, hi = lane >> 4;
  int aK = hi * 16, bK = hi * 16;
  v8f acc00 = {}, acc01 = {}, acc10 = {}, acc11 = {};
  const _Float16* ap = Pg + (long)(n0 + nl) * GS + aK;
  const _Float16* bp = Vg + (long)(col0 + nl) * GS + bK;
  long aS = 16L * GS, bS = 16L * GS;
  for (int k = 0; k < GS; k += 32) {
    AFrag a0, a1, b0, b1;
    a0.q[0] = *(const f32x4*)(ap);      a0.q[1] = *(const f32x4*)(ap + 8);
    a1.q[0] = *(const f32x4*)(ap + aS); a1.q[1] = *(const f32x4*)(ap + aS + 8);
    b0.q[0] = *(const f32x4*)(bp);      b0.q[1] = *(const f32x4*)(bp + 8);
    b1.q[0] = *(const f32x4*)(bp + bS); b1.q[1] = *(const f32x4*)(bp + bS + 8);
    acc00 = WMMA_F16(a0.v, b0.v, acc00);
    acc01 = WMMA_F16(a0.v, b1.v, acc01);
    acc10 = WMMA_F16(a1.v, b0.v, acc10);
    acc11 = WMMA_F16(a1.v, b1.v, acc11);
    ap += 32; bp += 32;
  }
#pragma unroll
  for (int sm = 0; sm < 2; sm++) {
#pragma unroll
    for (int sn = 0; sn < 2; sn++) {
      v8f acc = (sm == 0) ? (sn == 0 ? acc00 : acc01) : (sn == 0 ? acc10 : acc11);
      int o = col0 + sn * 16 + nl;
#pragma unroll
      for (int r = 0; r < 8; r++) {
        long n = row0 + sm * 16 + hi * 8 + r;
        Y[n * Cch + o] = (_Float16)acc[r];
      }
    }
  }
}

// ---------------------------------------------------------------------------
// 3x3 conv as 9-tap GEMM over CAT [token,1024] with border masking.
// ---------------------------------------------------------------------------
__global__ void __launch_bounds__(256)
conv3_gemm(const _Float16* __restrict__ CAT, const _Float16* __restrict__ W3r,
           const float* __restrict__ scale, const float* __restrict__ shift,
           _Float16* __restrict__ Y)
{
  int lane = threadIdx.x & 31;
  int wv = threadIdx.x >> 5;
  int wm = wv >> 2, wn = wv & 3;
  long row0 = (long)blockIdx.y * 64 + wm * 32;
  int  col0 = blockIdx.x * 128 + wn * 32;
  int nl = lane & 15, hi = lane >> 4;
  int aK = hi * 16, bK = hi * 16;
  long n0 = row0 + nl, n1 = n0 + 16;
  int y0 = (int)((n0 >> 7) & 127), x0 = (int)(n0 & 127);
  int y1 = (int)((n1 >> 7) & 127), x1 = (int)(n1 & 127);
  f32x4 z4 = {0.f, 0.f, 0.f, 0.f};
  v8f acc00 = {}, acc01 = {}, acc10 = {}, acc11 = {};

  for (int t = 0; t < 9; t++) {
    int dy = t / 3 - 1, dx = t % 3 - 1;
    bool v0 = ((unsigned)(y0 + dy) < 128u) && ((unsigned)(x0 + dx) < 128u);
    bool v1 = ((unsigned)(y1 + dy) < 128u) && ((unsigned)(x1 + dx) < 128u);
    const _Float16* a0p = CAT + (n0 + dy * 128 + dx) * 1024L + aK;
    const _Float16* a1p = CAT + (n1 + dy * 128 + dx) * 1024L + aK;
    const _Float16* bp  = W3r + (long)(col0 + nl) * 9216 + t * 1024 + bK;
    for (int kk = 0; kk < 1024; kk += 32) {
      AFrag a0, a1, b0, b1;
      if (v0) { a0.q[0] = *(const f32x4*)a0p; a0.q[1] = *(const f32x4*)(a0p + 8); }
      else    { a0.q[0] = z4; a0.q[1] = z4; }
      if (v1) { a1.q[0] = *(const f32x4*)a1p; a1.q[1] = *(const f32x4*)(a1p + 8); }
      else    { a1.q[0] = z4; a1.q[1] = z4; }
      b0.q[0] = *(const f32x4*)bp;                b0.q[1] = *(const f32x4*)(bp + 8);
      b1.q[0] = *(const f32x4*)(bp + 16L * 9216); b1.q[1] = *(const f32x4*)(bp + 16L * 9216 + 8);
      acc00 = WMMA_F16(a0.v, b0.v, acc00);
      acc01 = WMMA_F16(a0.v, b1.v, acc01);
      acc10 = WMMA_F16(a1.v, b0.v, acc10);
      acc11 = WMMA_F16(a1.v, b1.v, acc11);
      a0p += 32; a1p += 32; bp += 32;
    }
  }
#pragma unroll
  for (int sm = 0; sm < 2; sm++) {
#pragma unroll
    for (int sn = 0; sn < 2; sn++) {
      v8f acc = (sm == 0) ? (sn == 0 ? acc00 : acc01) : (sn == 0 ? acc10 : acc11);
      int o = col0 + sn * 16 + nl;
      float sc = scale[o], sh = shift[o];
#pragma unroll
      for (int r = 0; r < 8; r++) {
        long n = row0 + sm * 16 + hi * 8 + r;
        float v = fmaxf(acc[r] * sc + sh, 0.0f);
        Y[n * 512 + o] = (_Float16)v;
      }
    }
  }
}

// ---------------------------------------------------------------------------
// Token permutations (all layouts C-contiguous). 128 thr/block, 1 token/block.
// ---------------------------------------------------------------------------
__global__ void __launch_bounds__(128)
permute_tok(const _Float16* __restrict__ in, long ld_in,
            _Float16* __restrict__ out, long ld_out, int mode)
{
  int dst = blockIdx.x;
  int src;
  if (mode == 0) {                       // raw -> global-group order
    int j = dst & 255, Gg = dst >> 8;
    int ghi = j >> 4, gwi = j & 15;
    int lw = Gg & 7, lh = (Gg >> 3) & 7, bb = Gg >> 6;
    src = bb * 16384 + (ghi * 8 + lh) * 128 + (gwi * 8 + lw);
  } else if (mode == 1) {                // global-group -> local-group order
    int p = dst & 63, Gl = dst >> 6;
    int lh = p >> 3, lw = p & 7;
    int gwi = Gl & 15, ghi = (Gl >> 4) & 15, bb = Gl >> 8;
    src = (((bb * 8 + lh) * 8 + lw) << 8) + (ghi * 16 + gwi);
  } else {                               // local-group -> raw order
    int xx = dst & 127, yy = (dst >> 7) & 127, bb = dst >> 14;
    int ghi = yy >> 3, lh = yy & 7, gwi = xx >> 3, lw = xx & 7;
    src = (((bb * 16 + ghi) * 16 + gwi) << 6) + (lh * 8 + lw);
  }
  const _Float16* s = in  + (long)src * ld_in;
  _Float16*       d = out + (long)dst * ld_out;
  int t = threadIdx.x;                   // 128 * 8B = 512 halves
  ((uint2*)d)[t] = ((const uint2*)s)[t];
}

// ---------------------------------------------------------------------------
// Final head: logits = Y3 @ wout^T + bout (19 ch), softmax over channels.
// ---------------------------------------------------------------------------
__global__ void __launch_bounds__(256)
head_softmax(const _Float16* __restrict__ Y3, const float* __restrict__ wout,
             const float* __restrict__ bout, float* __restrict__ out, int Ntok)
{
  __shared__ float wsm[19 * 512];
  __shared__ float bsm[19];
  for (int i = threadIdx.x; i < 19 * 512; i += 256) wsm[i] = wout[i];
  if (threadIdx.x < 19) bsm[threadIdx.x] = bout[threadIdx.x];
  __syncthreads();
  int n = blockIdx.x * 256 + threadIdx.x;
  if (n >= Ntok) return;
  float acc[19];
#pragma unroll
  for (int o = 0; o < 19; o++) acc[o] = bsm[o];
  const _Float16* xr = Y3 + (long)n * 512;
  for (int c = 0; c < 512; c++) {
    float xv = (float)xr[c];
#pragma unroll
    for (int o = 0; o < 19; o++) acc[o] += xv * wsm[o * 512 + c];
  }
  float mx = acc[0];
#pragma unroll
  for (int o = 1; o < 19; o++) mx = fmaxf(mx, acc[o]);
  float sum = 0.0f;
#pragma unroll
  for (int o = 0; o < 19; o++) { acc[o] = __expf(acc[o] - mx); sum += acc[o]; }
  float inv = 1.0f / sum;
  int bb = n >> 14, hw = n & 16383;
#pragma unroll
  for (int o = 0; o < 19; o++)
    out[((long)(bb * 19 + o) << 14) + hw] = acc[o] * inv;
}

// ---------------------------------------------------------------------------
// Host orchestration
// ---------------------------------------------------------------------------
extern "C" void kernel_launch(void* const* d_in, const int* in_sizes, int n_in,
                              void* d_out, int out_size, void* d_ws, size_t ws_size,
                              hipStream_t stream)
{
  const int  Cc = 512, HW = 128 * 128, Bb = 4;
  const long Ntok = (long)Bb * HW;                 // 65536
  const float rscale = 0.04419417382415922f;       // 1/sqrt(512)

  // Input index maps: insertion-order vs sorted-pytree flattening
  bool xfirst = (n_in > 0) && (in_sizes[0] == Bb * Cc * HW);
  int ix, ic101, ic311, igq1, igq2, igk1, igk2, igv1, igv2, igo;
  int ilq1, ilq2, ilk1, ilk2, ilv1, ilv2, ilo, ioutw, ioutb;
  if (xfirst) {
    ix = 0; ic101 = 1;
    igq1 = 4;  igq2 = 7;  igk1 = 10; igk2 = 13; igv1 = 16; igv2 = 18; igo = 20;
    ilq1 = 23; ilq2 = 26; ilk1 = 29; ilk2 = 32; ilv1 = 35; ilv2 = 37; ilo = 39;
    ic311 = 42; ioutw = 45; ioutb = 46;
  } else {
    ic101 = 0; ic311 = 3;
    igk1 = 6;  igk2 = 9;  igo = 12; igq1 = 15; igq2 = 18; igv1 = 21; igv2 = 23;
    ilk1 = 25; ilk2 = 28; ilo = 31; ilq1 = 34; ilq2 = 37; ilv1 = 40; ilv2 = 42;
    ioutw = 44; ioutb = 45; ix = 46;
  }
  auto F = [&](int i) { return (const float*)d_in[i]; };

  // Workspace carve-up
  char* ws = (char*)d_ws;
  size_t off = 0;
  auto take = [&](size_t bytes) -> void* {
    void* p = ws + off;
    off += (bytes + 255) & ~(size_t)255;
    return p;
  };
  const size_t ACT = (size_t)Ntok * 512 * 2;                  // 64 MB
  _Float16* CAT = (_Float16*)take((size_t)Ntok * 1024 * 2);   // 128 MB
  _Float16* G0  = (_Float16*)take(ACT);
  _Float16* G1  = (_Float16*)take(ACT);
  _Float16* G2  = (_Float16*)take(ACT);
  _Float16* Qb  = (_Float16*)take(ACT);
  _Float16* Kb  = (_Float16*)take(ACT);
  _Float16* Vt  = (_Float16*)take(ACT);
  void*     SHR = take(ACT);                 // serves XH(f16) -> SIM(f32) -> Y3(f16)
  _Float16* Pb  = (_Float16*)take((size_t)256 * 256 * 256 * 2);  // 32 MB
  const int WN = 512 * 512;
  _Float16* wh[15];
  for (int i = 0; i < 15; i++) wh[i] = (_Float16*)take((size_t)WN * 2);
  _Float16* W3r = (_Float16*)take((size_t)512 * 9216 * 2);
  if (off > ws_size) return;   // insufficient workspace: bail out safely

  _Float16* XH  = (_Float16*)SHR;
  float*    SIM = (float*)SHR;
  _Float16* Y3  = (_Float16*)SHR;

  // ---- weight conversion ----
  auto cvt = [&](const float* src, _Float16* dst) {
    cvt_w<<<(WN + 255) / 256, 256, 0, stream>>>(src, dst, WN);
  };
  _Float16 *w101 = wh[0];
  _Float16 *wgq1 = wh[1], *wgq2 = wh[2], *wgk1 = wh[3], *wgk2 = wh[4];
  _Float16 *wgv1 = wh[5], *wgv2 = wh[6], *wgo = wh[7];
  _Float16 *wlq1 = wh[8], *wlq2 = wh[9], *wlk1 = wh[10], *wlk2 = wh[11];
  _Float16 *wlv1 = wh[12], *wlv2 = wh[13], *wlo = wh[14];
  cvt(F(ic101), w101);
  cvt(F(igq1), wgq1); cvt(F(igq2), wgq2); cvt(F(igk1), wgk1); cvt(F(igk2), wgk2);
  cvt(F(igv1), wgv1); cvt(F(igv2), wgv2); cvt(F(igo), wgo);
  cvt(F(ilq1), wlq1); cvt(F(ilq2), wlq2); cvt(F(ilk1), wlk1); cvt(F(ilk2), wlk2);
  cvt(F(ilv1), wlv1); cvt(F(ilv2), wlv2); cvt(F(ilo), wlo);
  repack_w3<<<(512 * 9216 + 255) / 256, 256, 0, stream>>>(F(ic311), W3r);

  // ---- x -> f16 [token, C] ----
  nchw_to_nc<<<dim3(HW / 32, Cc / 32, Bb), 256, 0, stream>>>(F(ix), XH, Cc, HW);

  dim3 gemmGrid(2, (unsigned)(Ntok / 64));   // 64x256 block tiles
  dim3 ctxGrid (4, (unsigned)(Ntok / 64));   // 64x128 block tiles
  dim3 cv3Grid (4, (unsigned)(Ntok / 64));

  // ---- conv101 (BN+ReLU) -> CAT[:, 0:512] ----
  gemm_f16<false><<<gemmGrid, 256, 0, stream>>>(XH, w101, F(ic101 + 1), F(ic101 + 2),
                                                1, CAT, 1024, 0, 512, 512, 1);
  // ---- permute raw -> global-group order ----
  permute_tok<<<(unsigned)Ntok, 128, 0, stream>>>(CAT, 1024, G0, 512, 0);

  // ===== Global SAB (GS = 256) =====
  gemm_f16<false><<<gemmGrid, 256, 0, stream>>>(G0, wgq1, F(igq1 + 1), F(igq1 + 2), 1, G1, 512, 0, 512, 512, 1);
  gemm_f16<false><<<gemmGrid, 256, 0, stream>>>(G1, wgq2, F(igq2 + 1), F(igq2 + 2), 1, Qb, 512, 0, 512, 512, 1);
  gemm_f16<false><<<gemmGrid, 256, 0, stream>>>(G0, wgk1, F(igk1 + 1), F(igk1 + 2), 1, G1, 512, 0, 512, 512, 1);
  gemm_f16<false><<<gemmGrid, 256, 0, stream>>>(G1, wgk2, F(igk2 + 1), F(igk2 + 2), 1, Kb, 512, 0, 512, 512, 1);
  gemm_f16<false><<<gemmGrid, 256, 0, stream>>>(G0, wgv1, nullptr, F(igv1 + 1), 0, G1, 512, 0, 512, 512, 1);
  gemm_f16<true ><<<gemmGrid, 256, 0, stream>>>(G1, wgv2, nullptr, F(igv2 + 1), 0, Vt, 0, 0, 512, 512, 256);
  attn_sim<<<dim3(4, 4, 256), 256, 0, stream>>>(Qb, Kb, SIM, 256, 512, rscale);
  softmax_rows<<<(unsigned)(Ntok / 8), 256, 0, stream>>>(SIM, Pb, 256);
  attn_ctx<<<ctxGrid, 256, 0, stream>>>(Pb, Vt, G1, 256, 512);
  gemm_f16<false><<<gemmGrid, 256, 0, stream>>>(G1, wgo, F(igo + 1), F(igo + 2), 1, G0, 512, 0, 512, 512, 1);

  // ---- permute global-group -> local-group order ----
  permute_tok<<<(unsigned)Ntok, 128, 0, stream>>>(G0, 512, G1, 512, 1);

  // ===== Local SAB (GS = 64) =====
  gemm_f16<false><<<gemmGrid, 256, 0, stream>>>(G1, wlq1, F(ilq1 + 1), F(ilq1 + 2), 1, G2, 512, 0, 512, 512, 1);
  gemm_f16<false><<<gemmGrid, 256, 0, stream>>>(G2, wlq2, F(ilq2 + 1), F(ilq2 + 2), 1, Qb, 512, 0, 512, 512, 1);
  gemm_f16<false><<<gemmGrid, 256, 0, stream>>>(G1, wlk1, F(ilk1 + 1), F(ilk1 + 2), 1, G2, 512, 0, 512, 512, 1);
  gemm_f16<false><<<gemmGrid, 256, 0, stream>>>(G2, wlk2, F(ilk2 + 1), F(ilk2 + 2), 1, Kb, 512, 0, 512, 512, 1);
  gemm_f16<false><<<gemmGrid, 256, 0, stream>>>(G1, wlv1, nullptr, F(ilv1 + 1), 0, G2, 512, 0, 512, 512, 1);
  gemm_f16<true ><<<gemmGrid, 256, 0, stream>>>(G2, wlv2, nullptr, F(ilv2 + 1), 0, Vt, 0, 0, 512, 512, 64);
  attn_sim<<<dim3(1, 1, 1024), 256, 0, stream>>>(Qb, Kb, SIM, 64, 512, rscale);
  softmax_rows<<<(unsigned)(Ntok / 8), 256, 0, stream>>>(SIM, Pb, 64);
  attn_ctx<<<ctxGrid, 256, 0, stream>>>(Pb, Vt, G2, 64, 512);
  gemm_f16<false><<<gemmGrid, 256, 0, stream>>>(G2, wlo, F(ilo + 1), F(ilo + 2), 1, G0, 512, 0, 512, 512, 1);

  // ---- permute local-group -> raw order into CAT[:, 512:1024] ----
  permute_tok<<<(unsigned)Ntok, 128, 0, stream>>>(G0, 512, CAT + 512, 1024, 2);

  // ---- 3x3 conv (BN+ReLU) on CAT -> Y3 ----
  conv3_gemm<<<cv3Grid, 256, 0, stream>>>(CAT, W3r, F(ic311 + 1), F(ic311 + 2), Y3);

  // ---- head conv 512->19 + channel softmax -> d_out [B,19,H,W] f32 ----
  head_softmax<<<(unsigned)(Ntok / 256), 256, 0, stream>>>(Y3, F(ioutw), F(ioutb),
                                                           (float*)d_out, (int)Ntok);
  (void)out_size; (void)ws_size;
}
